// LSTMPointerNet_32126355374094
// MI455X (gfx1250) — matrix operand order repaired and checked by reading, not compiled, not run on hardware
//
#include <hip/hip_runtime.h>

typedef __attribute__((ext_vector_type(16))) __bf16 v16bf;
typedef __attribute__((ext_vector_type(8)))  __bf16 v8bf;
typedef __attribute__((ext_vector_type(8)))  float  v8f;

#define B_   32
#define N_   128
#define D_   512
#define H_   512
#define T_   31
#define TQ   32
#define G4H  2048   // 4*H

// ---------- bf16 helpers: native casts (RNE) ----------
__device__ __forceinline__ __bf16 f2bf(float f) { return (__bf16)f; }
__device__ __forceinline__ float  bf2f(__bf16 b) { return (float)b; }
__device__ __forceinline__ void split_bf(float x, __bf16& hi, __bf16& lo) {
  hi = f2bf(x);
  lo = f2bf(x - bf2f(hi));
}

__device__ __forceinline__ float sigmoidf_(float x) {
  return 1.0f / (1.0f + __expf(-x));
}

// A-fragment: elements 0..7 <- p[0..7], 8..15 <- p[8..15] (16B-aligned b128 loads)
__device__ __forceinline__ v16bf ld_a(const __bf16* p) {
  v8bf lo = *(const v8bf*)p;
  v8bf hi = *(const v8bf*)(p + 8);
  return __builtin_shufflevector(lo, hi, 0, 1, 2, 3, 4, 5, 6, 7,
                                         8, 9, 10, 11, 12, 13, 14, 15);
}
// B-fragment (B pre-transposed to [N][K]): 16 contiguous bf16 (32B)
__device__ __forceinline__ v16bf ld_b(const __bf16* p) {
  return *(const v16bf*)p;
}

#define WMMA_BF16(A, Bf, Cc) \
  __builtin_amdgcn_wmma_f32_16x16x32_bf16(false, (A), false, (Bf), (short)0, (Cc), false, false)

// =====================================================================
// GEMM  C[M,N] = A[M,K] @ B[K,N], B staged transposed as Bt[N][K],
// A/B staged hi/lo bf16 (compensated: ah*bh + ah*bl + al*bh, fp32 acc).
// Per wave: 32x32 output tile (4 accumulators), 12 WMMA per 32-K step;
// every A/B fragment is reused twice -> ~1.5x less L2 fragment traffic
// per WMMA than a 16x32 tile. Optional Cadd[M,N] and bias[N].
// M%32==0, N%32==0, K%32==0.
// =====================================================================
__global__ void gemm_bf(const __bf16* __restrict__ Ahi, const __bf16* __restrict__ Alo,
                        const __bf16* __restrict__ Bhi, const __bf16* __restrict__ Blo,
                        const float* __restrict__ Cadd, const float* __restrict__ bias,
                        float* __restrict__ C, int M, int N, int K) {
  const int wave = threadIdx.x >> 5;
  const int lane = threadIdx.x & 31;
  const int tilesN = N >> 5;                 // 32-wide N tiles
  const int numTiles = (M >> 5) * tilesN;    // 32-tall M tiles
  const int tile = blockIdx.x * 8 + wave;
  if (tile >= numTiles) return;              // wave-uniform (EXEC all-1 for WMMA)
  const int tm = tile / tilesN, tn = tile % tilesN;
  const int half = lane >> 4, mr = lane & 15;

  v8f acc00 = {0.f, 0.f, 0.f, 0.f, 0.f, 0.f, 0.f, 0.f};
  v8f acc01 = acc00, acc10 = acc00, acc11 = acc00;

  const __bf16* A0h = Ahi + (long)(tm * 32 + mr) * K + half * 16;
  const __bf16* A0l = Alo + (long)(tm * 32 + mr) * K + half * 16;
  const __bf16* A1h = A0h + 16L * K;
  const __bf16* A1l = A0l + 16L * K;
  const __bf16* B0h = Bhi + (long)(tn * 32 + mr) * K + half * 16;
  const __bf16* B0l = Blo + (long)(tn * 32 + mr) * K + half * 16;
  const __bf16* B1h = B0h + 16L * K;
  const __bf16* B1l = B0l + 16L * K;

  for (int k0 = 0; k0 < K; k0 += 32) {
    v16bf a0h = ld_a(A0h + k0);
    v16bf a0l = ld_a(A0l + k0);
    v16bf a1h = ld_a(A1h + k0);
    v16bf a1l = ld_a(A1l + k0);
    v16bf b0h = ld_b(B0h + k0);
    v16bf b0l = ld_b(B0l + k0);
    v16bf b1h = ld_b(B1h + k0);
    v16bf b1l = ld_b(B1l + k0);
    acc00 = WMMA_BF16(a0h, b0h, acc00);
    acc00 = WMMA_BF16(a0h, b0l, acc00);
    acc00 = WMMA_BF16(a0l, b0h, acc00);
    acc01 = WMMA_BF16(a0h, b1h, acc01);
    acc01 = WMMA_BF16(a0h, b1l, acc01);
    acc01 = WMMA_BF16(a0l, b1h, acc01);
    acc10 = WMMA_BF16(a1h, b0h, acc10);
    acc10 = WMMA_BF16(a1h, b0l, acc10);
    acc10 = WMMA_BF16(a1l, b0h, acc10);
    acc11 = WMMA_BF16(a1h, b1h, acc11);
    acc11 = WMMA_BF16(a1h, b1l, acc11);
    acc11 = WMMA_BF16(a1l, b1h, acc11);
  }

#pragma unroll
  for (int vg = 0; vg < 8; ++vg) {
    long r0 = tm * 32 + half * 8 + vg;
    long r1 = r0 + 16;
    long c0 = tn * 32 + mr;
    long c1 = c0 + 16;
    float o00 = acc00[vg], o01 = acc01[vg], o10 = acc10[vg], o11 = acc11[vg];
    if (Cadd) {
      o00 += Cadd[r0 * N + c0]; o01 += Cadd[r0 * N + c1];
      o10 += Cadd[r1 * N + c0]; o11 += Cadd[r1 * N + c1];
    }
    if (bias) {
      float bc0 = bias[c0], bc1 = bias[c1];
      o00 += bc0; o01 += bc1; o10 += bc0; o11 += bc1;
    }
    C[r0 * N + c0] = o00; C[r0 * N + c1] = o01;
    C[r1 * N + c0] = o10; C[r1 * N + c1] = o11;
  }
}

// ---------- staging: f32 -> (hi,lo) bf16 ----------
__global__ void split_k(const float* __restrict__ src, __bf16* __restrict__ hi,
                        __bf16* __restrict__ lo, int n) {
  int idx = blockIdx.x * blockDim.x + threadIdx.x;
  if (idx >= n) return;
  split_bf(src[idx], hi[idx], lo[idx]);
}

// src [R][Ccols] -> dst [Ccols][R] with hi/lo split (weights [K,N] -> Bt[N,K])
__global__ void tsplit_k(const float* __restrict__ src, __bf16* __restrict__ hi,
                         __bf16* __restrict__ lo, int R, int Ccols) {
  int idx = blockIdx.x * blockDim.x + threadIdx.x;
  if (idx >= R * Ccols) return;
  int r = idx / Ccols, c = idx % Ccols;
  __bf16 h, l;
  split_bf(src[idx], h, l);
  hi[(long)c * R + r] = h;
  lo[(long)c * R + r] = l;
}

// ---------- LSTM prep ----------
__global__ void prep_xall(const float* __restrict__ lstm_in, const float* __restrict__ init_i,
                          __bf16* __restrict__ xhi, __bf16* __restrict__ xlo) {
  int idx = blockIdx.x * blockDim.x + threadIdx.x;      // TQ*B_*D_
  if (idx >= TQ * B_ * D_) return;
  int d = idx % D_;
  int r = idx / D_;                                     // r = t*B + b
  int t = r >> 5, b = r & 31;
  float x = (t == 0) ? init_i[d] : lstm_in[((long)b * T_ + (t - 1)) * D_ + d];
  split_bf(x, xhi[idx], xlo[idx]);
}

__global__ void init_hc(const float* __restrict__ init_h, const float* __restrict__ init_c,
                        __bf16* __restrict__ hhi, __bf16* __restrict__ hlo,
                        float* __restrict__ c) {
  int idx = blockIdx.x * blockDim.x + threadIdx.x;      // B_*H_
  if (idx >= B_ * H_) return;
  int j = idx % H_;
  split_bf(init_h[j], hhi[idx], hlo[idx]);
  c[idx] = init_c[j];
}

__global__ void bias_sum(const float* __restrict__ b_ih, const float* __restrict__ b_hh,
                         float* __restrict__ bsum) {
  int idx = blockIdx.x * blockDim.x + threadIdx.x;      // G4H
  if (idx >= G4H) return;
  bsum[idx] = b_ih[idx] + b_hh[idx];
}

// ---------- LSTM pointwise gate update (PyTorch order i,f,g,o) ----------
__global__ void lstm_step(const float* __restrict__ gates,
                          __bf16* __restrict__ hhi, __bf16* __restrict__ hlo,
                          float* __restrict__ c,
                          __bf16* __restrict__ qhi, __bf16* __restrict__ qlo, int t) {
  int idx = blockIdx.x * blockDim.x + threadIdx.x;      // B_*H_
  if (idx >= B_ * H_) return;
  int b = idx / H_, j = idx % H_;
  const float* g = gates + (long)b * G4H;
  float gi = g[j], gf = g[H_ + j], gg = g[2 * H_ + j], go = g[3 * H_ + j];
  float cn = sigmoidf_(gf) * c[idx] + sigmoidf_(gi) * tanhf(gg);
  float hn = sigmoidf_(go) * tanhf(cn);
  c[idx] = cn;
  __bf16 h, l;
  split_bf(hn, h, l);
  hhi[idx] = h; hlo[idx] = l;
  long q = ((long)b * TQ + t) * H_ + j;                 // query stored [B,Tq,H]
  qhi[q] = h; qlo[q] = l;
}

// =====================================================================
// Hop attention: score + mask + softmax + weighted sum -> newq (bf16 split).
// One 256-thread block per (b,t): 8 waves x 16 n-values; lanes stride H.
// =====================================================================
__global__ void attn_hop(const float* __restrict__ feat, const float* __restrict__ qp,
                         const float* __restrict__ v, const int* __restrict__ mem_sizes,
                         __bf16* __restrict__ outhi, __bf16* __restrict__ outlo) {
  __shared__ float sc[N_];
  __shared__ float red[256];
  const int bt = blockIdx.x;            // bt = b*TQ + t
  const int b = bt >> 5;
  const int wave = threadIdx.x >> 5, lane = threadIdx.x & 31;

  const float* qrow = qp + (long)bt * H_;
  float ql[16], vl[16];
#pragma unroll
  for (int j = 0; j < 16; ++j) { int hh = lane + 32 * j; ql[j] = qrow[hh]; vl[j] = v[hh]; }

  const float* fb = feat + (long)b * N_ * H_;
  for (int ni = 0; ni < 16; ++ni) {
    int n = wave * 16 + ni;
    const float* fr = fb + (long)n * H_ + lane;
    float acc = 0.f;
#pragma unroll
    for (int j = 0; j < 16; ++j) acc += vl[j] * tanhf(fr[32 * j] + ql[j]);
#pragma unroll
    for (int off = 16; off > 0; off >>= 1) acc += __shfl_xor(acc, off, 32);
    if (lane == 0) sc[n] = acc;
  }
  __syncthreads();

  const int msize = mem_sizes[b];
  float s = (threadIdx.x < N_) ? (((int)threadIdx.x < msize) ? sc[threadIdx.x] : -1e18f)
                               : -3.4e38f;
  red[threadIdx.x] = s; __syncthreads();
  for (int off = 128; off > 0; off >>= 1) {
    if ((int)threadIdx.x < off) red[threadIdx.x] = fmaxf(red[threadIdx.x], red[threadIdx.x + off]);
    __syncthreads();
  }
  float mx = red[0]; __syncthreads();
  float e = (threadIdx.x < N_) ? __expf(s - mx) : 0.f;
  red[threadIdx.x] = e; __syncthreads();
  for (int off = 128; off > 0; off >>= 1) {
    if ((int)threadIdx.x < off) red[threadIdx.x] += red[threadIdx.x + off];
    __syncthreads();
  }
  float inv = 1.0f / red[0]; __syncthreads();
  if (threadIdx.x < N_) sc[threadIdx.x] = e * inv;
  __syncthreads();

  for (int h0 = threadIdx.x; h0 < H_; h0 += 256) {
    float acc = 0.f;
    for (int n = 0; n < N_; ++n) acc += sc[n] * fb[(long)n * H_ + h0];
    long q = (long)bt * H_ + h0;
    split_bf(acc, outhi[q], outlo[q]);
  }
}

// ---------- Final raw attention score (no mask / no softmax) ----------
__global__ void attn_score(const float* __restrict__ feat, const float* __restrict__ qp,
                           const float* __restrict__ v, float* __restrict__ out) {
  const int bt = blockIdx.x;
  const int b = bt >> 5;
  const int wave = threadIdx.x >> 5, lane = threadIdx.x & 31;

  const float* qrow = qp + (long)bt * H_;
  float ql[16], vl[16];
#pragma unroll
  for (int j = 0; j < 16; ++j) { int hh = lane + 32 * j; ql[j] = qrow[hh]; vl[j] = v[hh]; }

  const float* fb = feat + (long)b * N_ * H_;
  for (int ni = 0; ni < 16; ++ni) {
    int n = wave * 16 + ni;
    const float* fr = fb + (long)n * H_ + lane;
    float acc = 0.f;
#pragma unroll
    for (int j = 0; j < 16; ++j) acc += vl[j] * tanhf(fr[32 * j] + ql[j]);
#pragma unroll
    for (int off = 16; off > 0; off >>= 1) acc += __shfl_xor(acc, off, 32);
    if (lane == 0) out[(long)bt * N_ + n] = acc;
  }
}

// =====================================================================
extern "C" void kernel_launch(void* const* d_in, const int* in_sizes, int n_in,
                              void* d_out, int out_size, void* d_ws, size_t ws_size,
                              hipStream_t stream) {
  (void)in_sizes; (void)n_in; (void)out_size; (void)ws_size;
  const float* attn_mem  = (const float*)d_in[0];
  const int*   mem_sizes = (const int*)  d_in[1];
  const float* lstm_in   = (const float*)d_in[2];
  const float* init_h    = (const float*)d_in[3];
  const float* init_c    = (const float*)d_in[4];
  const float* init_i    = (const float*)d_in[5];
  const float* w_ih      = (const float*)d_in[6];
  const float* w_hh      = (const float*)d_in[7];
  const float* b_ih      = (const float*)d_in[8];
  const float* b_hh      = (const float*)d_in[9];
  const float* attn_wm   = (const float*)d_in[10];
  const float* attn_wq   = (const float*)d_in[11];
  const float* attn_v    = (const float*)d_in[12];
  const float* hop_wm    = (const float*)d_in[13];
  const float* hop_wq    = (const float*)d_in[14];
  const float* hop_v     = (const float*)d_in[15];
  float* out = (float*)d_out;

  // ---- workspace carve: f32 region first, then bf16 region (all sizes are
  // ---- large powers of two -> every buffer stays >=32B aligned) ----
  float* wf = (float*)d_ws;
  float* attn_feat = wf; wf += (long)B_ * N_ * H_;    // 2.10M f32
  float* hop_feat  = wf; wf += (long)B_ * N_ * H_;    // 2.10M
  float* xpart     = wf; wf += (long)TQ * B_ * G4H;   // 2.10M
  float* bsum      = wf; wf += G4H;
  float* cbuf      = wf; wf += (long)B_ * H_;
  float* gates     = wf; wf += (long)B_ * G4H;
  float* qproj     = wf; wf += (long)B_ * TQ * H_;
  float* q2proj    = wf; wf += (long)B_ * TQ * H_;

  __bf16* wb = (__bf16*)wf;
  __bf16* am_hi  = wb; wb += (long)B_ * N_ * D_;      // attn_mem split
  __bf16* am_lo  = wb; wb += (long)B_ * N_ * D_;
  __bf16* x_hi   = wb; wb += (long)TQ * B_ * D_;      // Xall split
  __bf16* x_lo   = wb; wb += (long)TQ * B_ * D_;
  __bf16* h_hi   = wb; wb += (long)B_ * H_;
  __bf16* h_lo   = wb; wb += (long)B_ * H_;
  __bf16* q_hi   = wb; wb += (long)B_ * TQ * H_;      // query [B,Tq,H]
  __bf16* q_lo   = wb; wb += (long)B_ * TQ * H_;
  __bf16* nq_hi  = wb; wb += (long)B_ * TQ * H_;      // newq
  __bf16* nq_lo  = wb; wb += (long)B_ * TQ * H_;
  __bf16* wih_hi = wb; wb += (long)G4H * D_;          // w_ih already [N,K]
  __bf16* wih_lo = wb; wb += (long)G4H * D_;
  __bf16* whh_hi = wb; wb += (long)G4H * H_;
  __bf16* whh_lo = wb; wb += (long)G4H * H_;
  __bf16* awm_hi = wb; wb += (long)H_ * D_;           // attn_wm^T  [H][D]
  __bf16* awm_lo = wb; wb += (long)H_ * D_;
  __bf16* hwm_hi = wb; wb += (long)H_ * D_;           // hop_wm^T
  __bf16* hwm_lo = wb; wb += (long)H_ * D_;
  __bf16* hwq_hi = wb; wb += (long)H_ * H_;           // hop_wq^T
  __bf16* hwq_lo = wb; wb += (long)H_ * H_;
  __bf16* awq_hi = wb; wb += (long)H_ * H_;           // attn_wq^T
  __bf16* awq_lo = wb; wb += (long)H_ * H_;

  // ---- stage operands (one-time conversions) ----
  split_k<<<(B_ * N_ * D_) / 256, 256, 0, stream>>>(attn_mem, am_hi, am_lo, B_ * N_ * D_);
  split_k<<<(G4H * D_) / 256, 256, 0, stream>>>(w_ih, wih_hi, wih_lo, G4H * D_);
  split_k<<<(G4H * H_) / 256, 256, 0, stream>>>(w_hh, whh_hi, whh_lo, G4H * H_);
  tsplit_k<<<(D_ * H_) / 256, 256, 0, stream>>>(attn_wm, awm_hi, awm_lo, D_, H_);
  tsplit_k<<<(D_ * H_) / 256, 256, 0, stream>>>(hop_wm, hwm_hi, hwm_lo, D_, H_);
  tsplit_k<<<(H_ * H_) / 256, 256, 0, stream>>>(hop_wq, hwq_hi, hwq_lo, H_, H_);
  tsplit_k<<<(H_ * H_) / 256, 256, 0, stream>>>(attn_wq, awq_hi, awq_lo, H_, H_);
  prep_xall<<<(TQ * B_ * D_) / 256, 256, 0, stream>>>(lstm_in, init_i, x_hi, x_lo);
  init_hc<<<(B_ * H_) / 256, 256, 0, stream>>>(init_h, init_c, h_hi, h_lo, cbuf);
  bias_sum<<<G4H / 256, 256, 0, stream>>>(b_ih, b_hh, bsum);

  // ---- feature GEMMs: [4096,512]@[512,512]; tiles=(4096/32)*(512/32)=2048 -> 256 blocks
  gemm_bf<<<256, 256, 0, stream>>>(am_hi, am_lo, awm_hi, awm_lo, nullptr, nullptr,
                                   attn_feat, B_ * N_, H_, D_);
  gemm_bf<<<256, 256, 0, stream>>>(am_hi, am_lo, hwm_hi, hwm_lo, nullptr, nullptr,
                                   hop_feat, B_ * N_, H_, D_);

  // ---- time-parallel input transform: [1024,512]@[512,2048]; tiles=32*64=2048 -> 256 blocks
  gemm_bf<<<256, 256, 0, stream>>>(x_hi, x_lo, wih_hi, wih_lo, nullptr, nullptr,
                                   xpart, TQ * B_, G4H, D_);

  // ---- recurrent scan: gates = h@w_hh^T + xpart[t] + (b_ih+b_hh) ----
  for (int t = 0; t < TQ; ++t) {
    // M=32, N=2048 -> tiles = 1*64 = 64 -> 8 blocks
    gemm_bf<<<8, 256, 0, stream>>>(h_hi, h_lo, whh_hi, whh_lo,
                                   xpart + (long)t * B_ * G4H, bsum,
                                   gates, B_, G4H, H_);
    lstm_step<<<(B_ * H_) / 256, 256, 0, stream>>>(gates, h_hi, h_lo, cbuf,
                                                   q_hi, q_lo, t);
  }

  // ---- hop attention (masked softmax) ----
  // qproj = query @ hop_wq : tiles = 32*16 = 512 -> 64 blocks
  gemm_bf<<<64, 256, 0, stream>>>(q_hi, q_lo, hwq_hi, hwq_lo, nullptr, nullptr,
                                  qproj, B_ * TQ, H_, H_);
  attn_hop<<<B_ * TQ, 256, 0, stream>>>(hop_feat, qproj, hop_v, mem_sizes, nq_hi, nq_lo);

  // ---- final raw score [B,Tq,N] ----
  gemm_bf<<<64, 256, 0, stream>>>(nq_hi, nq_lo, awq_hi, awq_lo, nullptr, nullptr,
                                  q2proj, B_ * TQ, H_, H_);
  attn_score<<<B_ * TQ, 256, 0, stream>>>(attn_feat, q2proj, attn_v, out);
}
